// GPT2_1408749273890
// MI455X (gfx1250) — matrix operand order repaired and checked by reading, not compiled, not run on hardware
//
#include <hip/hip_runtime.h>
#include <math.h>

#define LAYERS 12
#define HEADS  12
#define CDIM   768
#define DHEAD  64
#define BATCH  8
#define SEQ    1024
#define VOCAB  50257
#define MTOK   (BATCH*SEQ)   // 8192 token rows

typedef __attribute__((ext_vector_type(16))) __bf16 bf16x16;
typedef __attribute__((ext_vector_type(8)))  __bf16 bf16x8;
typedef __attribute__((ext_vector_type(4)))  __bf16 bf16x4;
typedef __attribute__((ext_vector_type(8)))  float  f32x8;
typedef int i32x4v __attribute__((vector_size(16)));   // matches builtin param type

// ---- probe: gfx1250 async global->LDS copy ------------------------------
#if defined(__has_builtin)
#if __has_builtin(__builtin_amdgcn_global_load_async_to_lds_b128) && \
    __has_builtin(__builtin_amdgcn_s_wait_asynccnt)
#define USE_ASYNC_COPY 1
#endif
#endif
#ifndef USE_ASYNC_COPY
#define USE_ASYNC_COPY 0
#endif

#define GPTR1(p) ((__attribute__((address_space(1))) i32x4v*)(p))
#define LPTR3(p) ((__attribute__((address_space(3))) i32x4v*)(p))

// ---- fp32 -> bf16 (round-to-nearest-even) -------------------------------
static __device__ inline __bf16 f2bf(float f) {
    union { float f; unsigned u; } in; in.f = f;
    unsigned r = in.u + 0x7FFFu + ((in.u >> 16) & 1u);
    union { unsigned short s; __bf16 b; } out; out.s = (unsigned short)(r >> 16);
    return out.b;
}

// ---- WMMA wrapper (v_wmma_f32_16x16x32_bf16) ----------------------------
static __device__ inline f32x8 wmma_bf16(bf16x16 a, bf16x16 b, f32x8 c) {
    return __builtin_amdgcn_wmma_f32_16x16x32_bf16(false, a, false, b, (short)0, c, false, false);
}

// A fragment (16x32 bf16): lane row M=lane&15; element e -> K = 16*half + e
static __device__ inline bf16x16 load_a_frag(const __bf16* rowk, int half) {
    bf16x8 lo = *(const bf16x8*)(rowk + half * 16);
    bf16x8 hi = *(const bf16x8*)(rowk + half * 16 + 8);
    bf16x16 a;
#pragma unroll
    for (int i = 0; i < 8; ++i) { a[i] = lo[i]; a[i + 8] = hi[i]; }
    return a;
}
// B fragment (32x16 bf16): lane n = lane&15, element e -> K = 16*half + e.
// LDS tile [n][k] with padded k-stride (40 or 72) -> one aligned 32B read.

// ---- embedding ----------------------------------------------------------
__global__ __launch_bounds__(256)
void embed_kernel(const int* __restrict__ idx, const float* __restrict__ wte,
                  const float* __restrict__ wpe, float* __restrict__ x) {
    int m = blockIdx.x, t = threadIdx.x;
    int tok = idx[m];
    int pos = m & (SEQ - 1);
    const float* w = wte + (size_t)tok * CDIM;
    const float* p = wpe + (size_t)pos * CDIM;
    float* xr = x + (size_t)m * CDIM;
    xr[t]       = w[t]       + p[t];
    xr[t + 256] = w[t + 256] + p[t + 256];
    xr[t + 512] = w[t + 512] + p[t + 512];
}

// ---- layernorm (fp32 in, bf16 out), one block per row -------------------
__global__ __launch_bounds__(256)
void ln_kernel(const float* __restrict__ x, const float* __restrict__ g,
               const float* __restrict__ b, __bf16* __restrict__ out) {
    __shared__ float red[256];
    const int row = blockIdx.x, t = threadIdx.x;
    const float* xr = x + (size_t)row * CDIM;
    float v0 = xr[t], v1 = xr[t + 256], v2 = xr[t + 512];
    red[t] = v0 + v1 + v2;
    __syncthreads();
    for (int off = 128; off > 0; off >>= 1) { if (t < off) red[t] += red[t + off]; __syncthreads(); }
    float mean = red[0] * (1.0f / CDIM);
    __syncthreads();
    float d0 = v0 - mean, d1 = v1 - mean, d2 = v2 - mean;
    red[t] = d0 * d0 + d1 * d1 + d2 * d2;
    __syncthreads();
    for (int off = 128; off > 0; off >>= 1) { if (t < off) red[t] += red[t + off]; __syncthreads(); }
    float inv = rsqrtf(red[0] * (1.0f / CDIM) + 1e-5f);
    __bf16* orow = out + (size_t)row * CDIM;
    orow[t]       = f2bf(d0 * inv * g[t]       + b[t]);
    orow[t + 256] = f2bf(d1 * inv * g[t + 256] + b[t + 256]);
    orow[t + 512] = f2bf(d2 * inv * g[t + 512] + b[t + 512]);
}

// ---- final LN: last token per batch row; rows 8..15 zeroed --------------
__global__ __launch_bounds__(256)
void lnf_kernel(const float* __restrict__ x, const float* __restrict__ g,
                const float* __restrict__ b, __bf16* __restrict__ out) {
    __shared__ float red[256];
    const int row = blockIdx.x, t = threadIdx.x;
    __bf16* orow = out + (size_t)row * CDIM;
    if (row >= BATCH) {
        orow[t] = f2bf(0.f); orow[t + 256] = f2bf(0.f); orow[t + 512] = f2bf(0.f);
        return;
    }
    const float* xr = x + (size_t)(row * SEQ + SEQ - 1) * CDIM;
    float v0 = xr[t], v1 = xr[t + 256], v2 = xr[t + 512];
    red[t] = v0 + v1 + v2;
    __syncthreads();
    for (int off = 128; off > 0; off >>= 1) { if (t < off) red[t] += red[t + off]; __syncthreads(); }
    float mean = red[0] * (1.0f / CDIM);
    __syncthreads();
    float d0 = v0 - mean, d1 = v1 - mean, d2 = v2 - mean;
    red[t] = d0 * d0 + d1 * d1 + d2 * d2;
    __syncthreads();
    for (int off = 128; off > 0; off >>= 1) { if (t < off) red[t] += red[t + off]; __syncthreads(); }
    float inv = rsqrtf(red[0] * (1.0f / CDIM) + 1e-5f);
    orow[t]       = f2bf(d0 * inv * g[t]       + b[t]);
    orow[t + 256] = f2bf(d1 * inv * g[t + 256] + b[t + 256]);
    orow[t + 512] = f2bf(d2 * inv * g[t + 512] + b[t + 512]);
}

// ---- WMMA GEMM: Y[M,N] = epilogue(A_bf16[M,K] @ W_f32[K,N] + bias) ------
// EP 0: bf16 out; EP 1: bf16 out + exact GELU; EP 2: fp32 out + residual.
// 256 threads / 8 waves; block tile 128m x 64n; W tile (32k x 64n) staged
// bf16 in LDS as [n][k], k-stride 40 (80B rows): packed b64 stores,
// aligned b128 fragment reads spread across banks.
template <int EP>
__global__ __launch_bounds__(256)
void gemm_wmma(const __bf16* __restrict__ A, const float* __restrict__ W,
               const float* __restrict__ bias, int M, int N, int K,
               const float* __restrict__ res, float* __restrict__ outf,
               __bf16* __restrict__ outb) {
    __shared__ __align__(16) __bf16 lds_w[64 * 40];
    const int t = threadIdx.x;
    const int wid = t >> 5, lane = t & 31, half = lane >> 4, l15 = lane & 15;
    const int nb = blockIdx.x * 64, mb = blockIdx.y * 128;

    f32x8 acc[4];
#pragma unroll
    for (int nt = 0; nt < 4; ++nt)
#pragma unroll
        for (int r = 0; r < 8; ++r) acc[nt][r] = 0.f;

    const int arow = mb + wid * 16 + l15;
    for (int kb = 0; kb < K; kb += 32) {
        // stage 64n x 32k: 512 k-quads, 2 per thread; dword-coalesced global
        // reads along n, packed bf16x4 (b64) LDS stores along k.
#pragma unroll
        for (int i = 0; i < 2; ++i) {
            int j = i * 256 + t;
            int nn = j & 63, kq = j >> 6;              // kq in 0..7
            const float* wp = W + (size_t)(kb + kq * 4) * N + nb + nn;
            bf16x4 pk;
#pragma unroll
            for (int q = 0; q < 4; ++q) pk[q] = f2bf(wp[(size_t)q * N]);
            *(bf16x4*)(lds_w + nn * 40 + kq * 4) = pk;
            if (kb + 32 < K) __builtin_prefetch(wp + (size_t)32 * N, 0, 2);
        }
        __syncthreads();
        bf16x16 a = load_a_frag(A + (size_t)arow * K + kb, half);
#pragma unroll
        for (int nt = 0; nt < 4; ++nt) {
            bf16x16 bfr = *(const bf16x16*)(lds_w + (nt * 16 + l15) * 40 + half * 16);
            acc[nt] = wmma_bf16(a, bfr, acc[nt]);
        }
        __syncthreads();
    }
    const int orow = mb + wid * 16 + 8 * half;   // D layout: M = r + 8*half
#pragma unroll
    for (int nt = 0; nt < 4; ++nt) {
        int col = nb + nt * 16 + l15;            // D layout: N = lane&15
        float bv = bias[col];
#pragma unroll
        for (int r = 0; r < 8; ++r) {
            float v = acc[nt][r] + bv;
            size_t oi = (size_t)(orow + r) * N + col;
            if (EP == 0) {
                outb[oi] = f2bf(v);
            } else if (EP == 1) {
                v = 0.5f * v * (1.0f + erff(v * 0.70710678f));
                outb[oi] = f2bf(v);
            } else {
                outf[oi] = res[oi] + v;
            }
        }
    }
}

// ---- flash attention: one block = 64 query rows of one (batch, head) ----
__global__ __launch_bounds__(128)
void attn_kernel(const __bf16* __restrict__ qkv, __bf16* __restrict__ y) {
    __shared__ __align__(16) __bf16 lds_k[64 * 72];      // [key][d]
    __shared__ __align__(16) __bf16 lds_vt[64 * 72];     // [d][key]
    __shared__ __align__(16) __bf16 lds_p[4 * 16 * 72];  // per-wave [q][key]
    const int t = threadIdx.x, wid = t >> 5, lane = t & 31;
    const int half = lane >> 4, l15 = lane & 15;
    const int blk = blockIdx.x;
    const int qb = blk & 15;                 // SEQ/64 = 16
    const int hh = (blk >> 4) % HEADS;
    const int bb = blk / (16 * HEADS);
    const size_t rs = 3 * CDIM;

    const __bf16* qrow = qkv + (size_t)(bb * SEQ + qb * 64 + wid * 16 + l15) * rs + hh * DHEAD;
    bf16x16 qa0 = load_a_frag(qrow, half);
    bf16x16 qa1 = load_a_frag(qrow + 32, half);

    // this thread's 4 staged 16B chunks: chunk j -> key = j>>3, c8 = j&7
    int ckey[4], cc8[4];
    const __bf16* cbase[4];
#pragma unroll
    for (int i = 0; i < 4; ++i) {
        int j = i * 128 + t;
        ckey[i] = j >> 3; cc8[i] = j & 7;
    }

    f32x8 oacc[4];
    float mi[8], li[8];
#pragma unroll
    for (int nt = 0; nt < 4; ++nt)
#pragma unroll
        for (int r = 0; r < 8; ++r) oacc[nt][r] = 0.f;
#pragma unroll
    for (int r = 0; r < 8; ++r) { mi[r] = -1e30f; li[r] = 0.f; }

    for (int kb = 0; kb <= qb; ++kb) {
#pragma unroll
        for (int i = 0; i < 4; ++i)
            cbase[i] = qkv + (size_t)(bb * SEQ + kb * 64 + ckey[i]) * rs + hh * DHEAD + cc8[i] * 8;

        // ---- stage K [key][d] as b128; V transposed into [d][key] -------
        bf16x8 vc[4];
#if USE_ASYNC_COPY
#pragma unroll
        for (int i = 0; i < 4; ++i)
            __builtin_amdgcn_global_load_async_to_lds_b128(
                GPTR1(cbase[i] + CDIM),
                LPTR3(lds_k + ckey[i] * 72 + cc8[i] * 8),
                0, 0);
#else
        bf16x8 kc[4];
#pragma unroll
        for (int i = 0; i < 4; ++i) kc[i] = *(const bf16x8*)(cbase[i] + CDIM);
#endif
#pragma unroll
        for (int i = 0; i < 4; ++i) vc[i] = *(const bf16x8*)(cbase[i] + 2 * CDIM);
#if !USE_ASYNC_COPY
#pragma unroll
        for (int i = 0; i < 4; ++i)
            *(bf16x8*)(lds_k + ckey[i] * 72 + cc8[i] * 8) = kc[i];
#endif
#pragma unroll
        for (int i = 0; i < 4; ++i)
#pragma unroll
            for (int q = 0; q < 8; ++q)
                lds_vt[(cc8[i] * 8 + q) * 72 + ckey[i]] = vc[i][q];
        if (kb < qb)  // prefetch next K/V block
            __builtin_prefetch(cbase[0] + 64 * rs + CDIM, 0, 2);
#if USE_ASYNC_COPY
        __builtin_amdgcn_s_wait_asynccnt(0);
#endif
        __syncthreads();

        // ---- S = Q K^T * scale, causal mask -----------------------------
        float s[4][8];
#pragma unroll
        for (int nt = 0; nt < 4; ++nt) {
            f32x8 sa;
#pragma unroll
            for (int r = 0; r < 8; ++r) sa[r] = 0.f;
            const __bf16* kp = lds_k + (nt * 16 + l15) * 72;
            bf16x16 b0 = *(const bf16x16*)(kp + half * 16);
            bf16x16 b1 = *(const bf16x16*)(kp + 32 + half * 16);
            sa = wmma_bf16(qa0, b0, sa);
            sa = wmma_bf16(qa1, b1, sa);
            int keyg = kb * 64 + nt * 16 + l15;
#pragma unroll
            for (int r = 0; r < 8; ++r) {
                int qg = qb * 64 + wid * 16 + r + 8 * half;
                s[nt][r] = (keyg > qg) ? -1e30f : sa[r] * 0.125f;
            }
        }
        // ---- online softmax (row M lives on the 16 lanes of one half) ---
        float mnew[8];
#pragma unroll
        for (int r = 0; r < 8; ++r) {
            float pm = fmaxf(fmaxf(s[0][r], s[1][r]), fmaxf(s[2][r], s[3][r]));
            for (int off = 1; off < 16; off <<= 1)
                pm = fmaxf(pm, __shfl_xor(pm, off, 32));
            mnew[r] = fmaxf(mi[r], pm);
        }
#pragma unroll
        for (int r = 0; r < 8; ++r) {
            float alpha = __expf(mi[r] - mnew[r]);
            mi[r] = mnew[r];
            li[r] *= alpha;
#pragma unroll
            for (int nt = 0; nt < 4; ++nt) oacc[nt][r] *= alpha;
            float rsum = 0.f;
#pragma unroll
            for (int nt = 0; nt < 4; ++nt) {
                float pv = __expf(s[nt][r] - mnew[r]);
                s[nt][r] = pv;
                rsum += pv;
            }
            for (int off = 1; off < 16; off <<= 1) rsum += __shfl_xor(rsum, off, 32);
            li[r] += rsum;
        }
        // ---- P (D layout) -> LDS -> A layout; same-wave LDS is ordered --
        __bf16* pw = lds_p + wid * (16 * 72);
#pragma unroll
        for (int nt = 0; nt < 4; ++nt)
#pragma unroll
            for (int r = 0; r < 8; ++r)
                pw[(r + 8 * half) * 72 + nt * 16 + l15] = f2bf(s[nt][r]);
        // ---- O += P @ V -------------------------------------------------
#pragma unroll
        for (int ks = 0; ks < 2; ++ks) {
            bf16x16 pa = load_a_frag(pw + l15 * 72 + ks * 32, half);
#pragma unroll
            for (int nt = 0; nt < 4; ++nt) {
                bf16x16 vb = *(const bf16x16*)(lds_vt + (nt * 16 + l15) * 72 + ks * 32 + half * 16);
                oacc[nt] = wmma_bf16(pa, vb, oacc[nt]);
            }
        }
        __syncthreads();
    }
#pragma unroll
    for (int nt = 0; nt < 4; ++nt)
#pragma unroll
        for (int r = 0; r < 8; ++r) {
            int trow = bb * SEQ + qb * 64 + wid * 16 + r + 8 * half;
            y[(size_t)trow * CDIM + hh * DHEAD + nt * 16 + l15] = f2bf(oacc[nt][r] / li[r]);
        }
}

// ---- logits: [16(8 used) x 768] @ wte^T -> [8, V], WMMA -----------------
__global__ __launch_bounds__(128)
void logits_kernel(const __bf16* __restrict__ xf, const float* __restrict__ wte,
                   float* __restrict__ out) {
    __shared__ __align__(16) __bf16 lds_w[64 * 40];      // [v][c]
    const int t = threadIdx.x, wid = t >> 5, lane = t & 31;
    const int half = lane >> 4, l15 = lane & 15;
    const int vb = blockIdx.x * 64;
    f32x8 acc;
#pragma unroll
    for (int r = 0; r < 8; ++r) acc[r] = 0.f;

    typedef __attribute__((ext_vector_type(4))) float f32x4;
    for (int kb = 0; kb < CDIM; kb += 32) {
        // 64v x 32c in float4 chunks: 512 quads, 4 per thread; wte rows are
        // contiguous along c, so both loads and b64 LDS stores are vector.
#pragma unroll
        for (int i = 0; i < 4; ++i) {
            int j = i * 128 + t;
            int vv = j >> 3, c4 = (j & 7) * 4;
            int vg = vb + vv;
            bf16x4 pk;
            if (vg < VOCAB) {
                f32x4 w = *(const f32x4*)(wte + (size_t)vg * CDIM + kb + c4);
#pragma unroll
                for (int q = 0; q < 4; ++q) pk[q] = f2bf(w[q]);
            } else {
#pragma unroll
                for (int q = 0; q < 4; ++q) pk[q] = f2bf(0.f);
            }
            *(bf16x4*)(lds_w + vv * 40 + c4) = pk;
        }
        __syncthreads();
        bf16x16 a = load_a_frag(xf + (size_t)l15 * CDIM + kb, half);
        bf16x16 b = *(const bf16x16*)(lds_w + (wid * 16 + l15) * 40 + half * 16);
        acc = wmma_bf16(a, b, acc);
        __syncthreads();
    }
#pragma unroll
    for (int r = 0; r < 8; ++r) {
        int bidx = r + 8 * half;
        int vcol = vb + wid * 16 + l15;
        if (bidx < BATCH && vcol < VOCAB)
            out[(size_t)bidx * VOCAB + vcol] = acc[r];
    }
}

// -------------------------------------------------------------------------
extern "C" void kernel_launch(void* const* d_in, const int* in_sizes, int n_in,
                              void* d_out, int out_size, void* d_ws, size_t ws_size,
                              hipStream_t stream) {
    const int*   idx  = (const int*)d_in[0];
    const float* wte  = (const float*)d_in[1];
    const float* wpe  = (const float*)d_in[2];
    const float* ln1g = (const float*)d_in[3];
    const float* ln1b = (const float*)d_in[4];
    const float* wqkv = (const float*)d_in[5];
    const float* bqkv = (const float*)d_in[6];
    const float* wao  = (const float*)d_in[7];
    const float* bao  = (const float*)d_in[8];
    const float* ln2g = (const float*)d_in[9];
    const float* ln2b = (const float*)d_in[10];
    const float* wfc  = (const float*)d_in[11];
    const float* bfc  = (const float*)d_in[12];
    const float* wmo  = (const float*)d_in[13];
    const float* bmo  = (const float*)d_in[14];
    const float* lnfg = (const float*)d_in[15];
    const float* lnfb = (const float*)d_in[16];
    float* out = (float*)d_out;
    (void)in_sizes; (void)n_in; (void)out_size; (void)ws_size;

    char* ws = (char*)d_ws;
    size_t off = 0;
    auto alloc = [&](size_t bytes) {
        char* p = ws + off;
        off = (off + bytes + 255) & ~(size_t)255;
        return p;
    };
    float*  x    = (float*) alloc((size_t)MTOK * CDIM * 4);        // residual stream
    __bf16* hbuf = (__bf16*)alloc((size_t)MTOK * CDIM * 2);        // LN output
    __bf16* big  = (__bf16*)alloc((size_t)MTOK * 4 * CDIM * 2);    // qkv / fc-act (disjoint lifetimes)
    __bf16* ybuf = (__bf16*)alloc((size_t)MTOK * CDIM * 2);        // attention out
    __bf16* xf16 = (__bf16*)alloc((size_t)16 * CDIM * 2);          // final-token LN (padded 16 rows)
    __bf16* qkvb = big;   // [M, 3C]
    __bf16* gbuf = big;   // [M, 4C]

    embed_kernel<<<MTOK, 256, 0, stream>>>(idx, wte, wpe, x);

    for (int l = 0; l < LAYERS; ++l) {
        ln_kernel<<<MTOK, 256, 0, stream>>>(x, ln1g + l * CDIM, ln1b + l * CDIM, hbuf);
        gemm_wmma<0><<<dim3(3 * CDIM / 64, MTOK / 128), 256, 0, stream>>>(
            hbuf, wqkv + (size_t)l * CDIM * 3 * CDIM, bqkv + l * 3 * CDIM,
            MTOK, 3 * CDIM, CDIM, nullptr, nullptr, qkvb);
        attn_kernel<<<BATCH * HEADS * (SEQ / 64), 128, 0, stream>>>(qkvb, ybuf);
        gemm_wmma<2><<<dim3(CDIM / 64, MTOK / 128), 256, 0, stream>>>(
            ybuf, wao + (size_t)l * CDIM * CDIM, bao + l * CDIM,
            MTOK, CDIM, CDIM, x, x, nullptr);
        ln_kernel<<<MTOK, 256, 0, stream>>>(x, ln2g + l * CDIM, ln2b + l * CDIM, hbuf);
        gemm_wmma<1><<<dim3(4 * CDIM / 64, MTOK / 128), 256, 0, stream>>>(
            hbuf, wfc + (size_t)l * CDIM * 4 * CDIM, bfc + l * 4 * CDIM,
            MTOK, 4 * CDIM, CDIM, nullptr, nullptr, gbuf);
        gemm_wmma<2><<<dim3(CDIM / 64, MTOK / 128), 256, 0, stream>>>(
            gbuf, wmo + (size_t)l * 4 * CDIM * CDIM, bmo + l * CDIM,
            MTOK, CDIM, 4 * CDIM, x, x, nullptr);
    }

    lnf_kernel<<<16, 256, 0, stream>>>(x, lnfg, lnfb, xf16);
    logits_kernel<<<(VOCAB + 63) / 64, 128, 0, stream>>>(xf16, wte, out);
}
